// AttentionRecognitionHead_12713103196987
// MI455X (gfx1250) — matrix operand order repaired and checked by reading, not compiled, not run on hardware
//
#include <hip/hip_runtime.h>
#include <hip/hip_bf16.h>

typedef _Float16 half_t;
typedef __attribute__((ext_vector_type(16))) _Float16 v16h;
typedef __attribute__((ext_vector_type(8)))  _Float16 v8h;
typedef __attribute__((ext_vector_type(8)))  float    v8f;
typedef __attribute__((ext_vector_type(4)))  float    v4f;
typedef int v4i_t __attribute__((vector_size(16)));   // 4 x i32, matches builtin proto

#define B_  256
#define T_  1024
#define DX_ 512
#define S_  512
#define A_  512
#define C_  97
#define L_  25
#define BT_ (B_*T_)

// attention LDS chunking
#define ROWS_   64                 // rows (t) per chunk
#define RPAD_   520                // padded row stride in halves (512 + 8)
#define CHUNKS_ (T_ / ROWS_)       // 16
#define LPT_    16                 // b128 transfers per thread per chunk (64*64/256)
#define XBUF_HALVES (2 * ROWS_ * RPAD_)

// ---------------- CDNA5 async global->LDS copy path ----------------
#if defined(__has_builtin)
#  if __has_builtin(__builtin_amdgcn_global_load_async_to_lds_b128)
#    define ASYNC_LDS 1
#  endif
#endif

typedef __attribute__((address_space(1))) v4i_t* gv4i_p;   // global 16B vector ptr
typedef __attribute__((address_space(3))) v4i_t* lv4i_p;   // LDS 16B vector ptr

__device__ __forceinline__ void async_copy16(const half_t* __restrict__ g,
                                             half_t* __restrict__ l) {
#ifdef ASYNC_LDS
  __builtin_amdgcn_global_load_async_to_lds_b128(
      (gv4i_p)(half_t*)g, (lv4i_p)l, 0, 0);
#else
  *(v8h*)l = *(const v8h*)g;   // fallback: load -> ds_store through same pipeline
#endif
}

__device__ __forceinline__ void async_wait_pending() {
#ifdef ASYNC_LDS
  asm volatile("s_wait_asynccnt 0x10" ::: "memory");  // <=16 outstanding: prev chunk done
#endif
}
__device__ __forceinline__ void async_wait_all() {
#ifdef ASYNC_LDS
  asm volatile("s_wait_asynccnt 0x0" ::: "memory");
#endif
}

// ---------------- elementwise helpers ----------------

__global__ void cvt_f32_f16(const float* __restrict__ in, half_t* __restrict__ out, long long n4) {
  long long i = (long long)blockIdx.x * blockDim.x + threadIdx.x;
  if (i >= n4) return;
  v4f v = *(const v4f*)(in + i * 4);
  half_t* o = out + i * 4;
  o[0] = (half_t)v.x; o[1] = (half_t)v.y; o[2] = (half_t)v.z; o[3] = (half_t)v.w;
}

__global__ void zero_state(float* __restrict__ h, half_t* __restrict__ hh, int n) {
  int i = blockIdx.x * blockDim.x + threadIdx.x;
  if (i < n) { h[i] = 0.f; hh[i] = (half_t)0.f; }
}

// ---------------- WMMA GEMM: out[m,n] = sum_k A[m,k] * W[n,k] + bias[n] ----------------
// A: fp16 row-major [M,K]; W: fp16 row-major [N,K] (torch weight layout).
// Each wave32 computes a 16(M) x 64(N) tile: 1 A-frag reused across 4 B-frags
// -> 4 v_wmma_f32_16x16x32_f16 per K-chunk of 32.

__device__ __forceinline__ v16h load_a_frag(const half_t* __restrict__ Abase, int ldk,
                                            int lane, int k0) {
  // A fragment (contiguous k order, same as B): lane%16 = M row; lanes 0-15 K=0..15, lanes 16-31 K=16..31.
  int m  = lane & 15;
  int hs = lane >> 4;
  const half_t* p = Abase + (long long)m * ldk + k0 + hs * 16;
  v8h lo = *(const v8h*)(p);        // K = hs*16 .. hs*16+7
  v8h hi = *(const v8h*)(p + 8);    // K = hs*16 + 8 .. +7
  v16h r;
#pragma unroll
  for (int i = 0; i < 8; ++i) { r[i] = lo[i]; r[i + 8] = hi[i]; }
  return r;
}

__device__ __forceinline__ v16h load_b_frag(const half_t* __restrict__ W, int ldk,
                                            int lane, int n0, int k0) {
  // B-matrix 32x16: lane%16 = N column; lanes 0-15 K=0..15, lanes 16-31 K=16..31.
  int n  = n0 + (lane & 15);
  int hs = lane >> 4;
  return *(const v16h*)(W + (long long)n * ldk + k0 + hs * 16);
}

__global__ void gemm_wmma(const half_t* __restrict__ Ain, const half_t* __restrict__ W,
                          const float* __restrict__ bias,
                          float* __restrict__ outF, half_t* __restrict__ outH,
                          int M, int N, int K) {
  int wavesPerBlock = blockDim.x >> 5;
  int wave = blockIdx.x * wavesPerBlock + ((int)threadIdx.x >> 5);
  int lane = (int)threadIdx.x & 31;
  int tilesN = N >> 6;              // 64-wide N tiles
  int tilesM = M >> 4;
  if (wave >= tilesM * tilesN) return;   // whole-wave uniform exit (EXEC all-1s for WMMA)
  int tm = wave / tilesN;
  int tn = wave - tm * tilesN;

  const half_t* Abase = Ain + (long long)tm * 16 * K;
  int nbase = tn << 6;
  v8f acc0 = {}, acc1 = {}, acc2 = {}, acc3 = {};
  for (int k0 = 0; k0 < K; k0 += 32) {
    v16h a  = load_a_frag(Abase, K, lane, k0);
    v16h b0 = load_b_frag(W, K, lane, nbase,      k0);
    v16h b1 = load_b_frag(W, K, lane, nbase + 16, k0);
    v16h b2 = load_b_frag(W, K, lane, nbase + 32, k0);
    v16h b3 = load_b_frag(W, K, lane, nbase + 48, k0);
    acc0 = __builtin_amdgcn_wmma_f32_16x16x32_f16(false, a, false, b0, (short)0, acc0, false, false);
    acc1 = __builtin_amdgcn_wmma_f32_16x16x32_f16(false, a, false, b1, (short)0, acc1, false, false);
    acc2 = __builtin_amdgcn_wmma_f32_16x16x32_f16(false, a, false, b2, (short)0, acc2, false, false);
    acc3 = __builtin_amdgcn_wmma_f32_16x16x32_f16(false, a, false, b3, (short)0, acc3, false, false);
  }

  // D layout: VGPR j -> row = (lane/16)*8 + j, col = lane%16
  int mrow = (tm << 4) + ((lane >> 4) << 3);
  v8f accs[4] = {acc0, acc1, acc2, acc3};
#pragma unroll
  for (int s = 0; s < 4; ++s) {
    int n = nbase + (s << 4) + (lane & 15);
    float bv = bias ? bias[n] : 0.f;
#pragma unroll
    for (int j = 0; j < 8; ++j) {
      float v = accs[s][j] + bv;
      long long idx = (long long)(mrow + j) * N + n;
      if (outF) outF[idx] = v;
      if (outH) outH[idx] = (half_t)v;
    }
  }
}

// ---------------- fused attention step (one block of 256 per batch row) ----------------
// scores[t] = sum_a tanh(sProj[b,a] + xProj[b,t,a]) * wW[a] + wb ; alpha = softmax(scores)
// context[d] = sum_t alpha[t] * x[b,t,d] ; gin[b] = [emb[y_prev], context] (fp16)
// xProj / x rows are streamed through double-buffered LDS chunks filled with
// async global->LDS copies (ASYNCcnt) while the previous chunk is consumed.

__global__ void attn_step(const half_t* __restrict__ xProjh, const half_t* __restrict__ xh,
                          const float* __restrict__ sProj,
                          const float* __restrict__ wW, const float* __restrict__ wb,
                          const float* __restrict__ emb, const int* __restrict__ targets,
                          int step, half_t* __restrict__ gin) {
  extern __shared__ char smem[];
  half_t* XB  = (half_t*)smem;                         // 2 x ROWS_ x RPAD_ halves
  float*  sc  = (float*)(smem + XBUF_HALVES * sizeof(half_t));   // [T_]
  float*  sP  = sc + T_;                               // [A_]
  float*  sw  = sP + A_;                               // [A_]
  float*  red = sw + A_;                               // [256]

  int b   = blockIdx.x;
  int tid = threadIdx.x;

  for (int a = tid; a < A_; a += 256) { sP[a] = sProj[b * A_ + a]; sw[a] = wW[a]; }
  __syncthreads();

  const half_t* xp = xProjh + (long long)b * T_ * A_;
  const half_t* xr = xh     + (long long)b * T_ * DX_;
  float wb0 = wb[0];

  // ---- issue helper: copy 64 rows (chunk c of src) into buffer buf ----
  // 4096 x 16B transfers per chunk -> 16 per thread.
  auto issue_chunk = [&](const half_t* src, int c, int buf) {
    half_t* base = XB + buf * (ROWS_ * RPAD_);
    const half_t* gsrc = src + (long long)c * ROWS_ * 512;
#pragma unroll
    for (int i = 0; i < LPT_; ++i) {
      int idx = tid + i * 256;          // 0..4095
      int row = idx >> 6;               // 0..63
      int seg = idx & 63;               // 0..63 (16B segments)
      async_copy16(gsrc + row * 512 + seg * 8, base + row * RPAD_ + seg * 8);
    }
  };

  // ================= Phase A: scores =================
  issue_chunk(xp, 0, 0);
  int rowl = tid >> 2;      // 0..63: row within chunk (4 lanes per row)
  int q    = tid & 3;       // quarter of the a-dimension
  for (int c = 0; c < CHUNKS_; ++c) {
    if (c + 1 < CHUNKS_) { issue_chunk(xp, c + 1, (c + 1) & 1); async_wait_pending(); }
    else                 { async_wait_all(); }
    __syncthreads();
    const half_t* rp = XB + (c & 1) * (ROWS_ * RPAD_) + rowl * RPAD_ + q * 128;
    float s = 0.f;
    for (int a = 0; a < 128; a += 8) {
      v8h v = *(const v8h*)(rp + a);
      int ab = q * 128 + a;
#pragma unroll
      for (int i = 0; i < 8; ++i) s += tanhf(sP[ab + i] + (float)v[i]) * sw[ab + i];
    }
    // reduce the 4 quarter-sums within 4 adjacent lanes (wave32 shuffle)
    s += __shfl_xor(s, 1, 32);
    s += __shfl_xor(s, 2, 32);
    if (q == 0) sc[c * ROWS_ + rowl] = s + wb0;
    __syncthreads();
  }

  // ================= softmax over T in LDS =================
  float lmax = -3.4e38f;
  for (int t = tid; t < T_; t += 256) lmax = fmaxf(lmax, sc[t]);
  red[tid] = lmax; __syncthreads();
  for (int s2 = 128; s2 > 0; s2 >>= 1) {
    if (tid < s2) red[tid] = fmaxf(red[tid], red[tid + s2]);
    __syncthreads();
  }
  float mx = red[0]; __syncthreads();

  float lsum = 0.f;
  for (int t = tid; t < T_; t += 256) { float e = __expf(sc[t] - mx); sc[t] = e; lsum += e; }
  red[tid] = lsum; __syncthreads();
  for (int s2 = 128; s2 > 0; s2 >>= 1) {
    if (tid < s2) red[tid] += red[tid + s2];
    __syncthreads();
  }
  float inv = 1.f / red[0]; __syncthreads();
  for (int t = tid; t < T_; t += 256) sc[t] *= inv;
  __syncthreads();

  // ================= Phase B: context =================
  float c0 = 0.f, c1 = 0.f;           // thread owns d = tid and d = tid + 256
  issue_chunk(xr, 0, 0);
  for (int c = 0; c < CHUNKS_; ++c) {
    if (c + 1 < CHUNKS_) { issue_chunk(xr, c + 1, (c + 1) & 1); async_wait_pending(); }
    else                 { async_wait_all(); }
    __syncthreads();
    const half_t* base = XB + (c & 1) * (ROWS_ * RPAD_);
    const float*  al   = sc + c * ROWS_;
#pragma unroll 4
    for (int r = 0; r < ROWS_; ++r) {
      float a = al[r];
      c0 += a * (float)base[r * RPAD_ + tid];
      c1 += a * (float)base[r * RPAD_ + 256 + tid];
    }
    __syncthreads();
  }
  half_t* gb = gin + (long long)b * (A_ + DX_);
  gb[A_ + tid]       = (half_t)c0;
  gb[A_ + 256 + tid] = (half_t)c1;

  // teacher-forced embedding: step 0 uses <start> id == C_, else targets[:, step-1]
  int yid = (step == 0) ? C_ : targets[b * L_ + step - 1];
  for (int a = tid; a < A_; a += 256)
    gb[a] = (half_t)emb[(long long)yid * A_ + a];
}

// ---------------- GRU gate fusion ----------------

__global__ void gru_gate(const float* __restrict__ gi, const float* __restrict__ gh,
                         float* __restrict__ h, half_t* __restrict__ hh) {
  int i = blockIdx.x * blockDim.x + threadIdx.x;
  if (i >= B_ * S_) return;
  int b = i / S_, j = i - b * S_;
  const float* gib = gi + (long long)b * 3 * S_;
  const float* ghb = gh + (long long)b * 3 * S_;
  float ir = gib[j],           hr = ghb[j];
  float iz = gib[S_ + j],      hz = ghb[S_ + j];
  float in_ = gib[2 * S_ + j], hn = ghb[2 * S_ + j];
  float r = 1.f / (1.f + __expf(-(ir + hr)));
  float z = 1.f / (1.f + __expf(-(iz + hz)));
  float n = tanhf(in_ + r * hn);
  float hprev = h[i];
  float hnew = (1.f - z) * n + z * hprev;
  h[i] = hnew;
  hh[i] = (half_t)hnew;
}

// ---------------- output fc (C=97, ragged; plain VALU) ----------------

__global__ void fc_out(const float* __restrict__ h, const float* __restrict__ fcW,
                       const float* __restrict__ fcb, float* __restrict__ out, int step) {
  int i = blockIdx.x * blockDim.x + threadIdx.x;
  if (i >= B_ * C_) return;
  int b = i / C_, c = i - b * C_;
  const float* hr = h + (long long)b * S_;
  const float* w  = fcW + (long long)c * S_;
  float s = fcb[c];
  for (int k = 0; k < S_; ++k) s += hr[k] * w[k];
  out[((long long)b * L_ + step) * C_ + c] = s;
}

// ---------------- host launcher ----------------

extern "C" void kernel_launch(void* const* d_in, const int* in_sizes, int n_in,
                              void* d_out, int out_size, void* d_ws, size_t ws_size,
                              hipStream_t stream) {
  (void)in_sizes; (void)n_in; (void)out_size; (void)ws_size;
  const float* x     = (const float*)d_in[0];
  const int*   tgt   = (const int*)d_in[1];
  /* d_in[2] lengths: unused (reference discards) */
  const float* sW    = (const float*)d_in[3];
  const float* sb    = (const float*)d_in[4];
  const float* xW    = (const float*)d_in[5];
  const float* xb    = (const float*)d_in[6];
  const float* wW    = (const float*)d_in[7];
  const float* wb    = (const float*)d_in[8];
  const float* emb   = (const float*)d_in[9];
  const float* W_ih  = (const float*)d_in[10];
  const float* W_hh  = (const float*)d_in[11];
  const float* b_ih  = (const float*)d_in[12];
  const float* b_hh  = (const float*)d_in[13];
  const float* fcW   = (const float*)d_in[14];
  const float* fcb   = (const float*)d_in[15];
  float* out = (float*)d_out;

  char* ws = (char*)d_ws;
  size_t off = 0;
  auto carve = [&](size_t bytes) -> void* {
    void* p = ws + off;
    off = (off + bytes + 255) & ~(size_t)255;
    return p;
  };
  half_t* xh    = (half_t*)carve((size_t)BT_ * DX_ * sizeof(half_t));
  half_t* xPh   = (half_t*)carve((size_t)BT_ * A_ * sizeof(half_t));
  half_t* xWh   = (half_t*)carve((size_t)A_ * DX_ * sizeof(half_t));
  half_t* sWh   = (half_t*)carve((size_t)A_ * S_ * sizeof(half_t));
  half_t* Wih_h = (half_t*)carve((size_t)3 * S_ * (DX_ + A_) * sizeof(half_t));
  half_t* Whh_h = (half_t*)carve((size_t)3 * S_ * S_ * sizeof(half_t));
  float*  h     = (float*)carve((size_t)B_ * S_ * sizeof(float));
  half_t* hh    = (half_t*)carve((size_t)B_ * S_ * sizeof(half_t));
  float*  sProj = (float*)carve((size_t)B_ * A_ * sizeof(float));
  half_t* gin   = (half_t*)carve((size_t)B_ * (A_ + DX_) * sizeof(half_t));
  float*  gi    = (float*)carve((size_t)B_ * 3 * S_ * sizeof(float));
  float*  gh    = (float*)carve((size_t)B_ * 3 * S_ * sizeof(float));

  auto cvt = [&](const float* src, half_t* dst, long long n) {
    long long n4 = n >> 2;
    int blocks = (int)((n4 + 255) / 256);
    cvt_f32_f16<<<blocks, 256, 0, stream>>>(src, dst, n4);
  };
  cvt(x,    xh,    (long long)BT_ * DX_);
  cvt(xW,   xWh,   (long long)A_ * DX_);
  cvt(sW,   sWh,   (long long)A_ * S_);
  cvt(W_ih, Wih_h, (long long)3 * S_ * (DX_ + A_));
  cvt(W_hh, Whh_h, (long long)3 * S_ * S_);

  zero_state<<<(B_ * S_ + 255) / 256, 256, 0, stream>>>(h, hh, B_ * S_);

  auto gemm = [&](const half_t* Ain, const half_t* Wm, const float* bias,
                  float* oF, half_t* oH, int M, int N, int K) {
    int waves  = (M >> 4) * (N >> 6);     // 16x64 tile per wave
    int blocks = (waves + 7) / 8;          // 8 wave32s per 256-thread block
    gemm_wmma<<<blocks, 256, 0, stream>>>(Ain, Wm, bias, oF, oH, M, N, K);
  };

  // hoisted xProj = x @ xW^T + xb, stored fp16 to halve per-step stream bytes
  gemm(xh, xWh, xb, nullptr, xPh, BT_, A_, DX_);

  size_t attn_lds = XBUF_HALVES * sizeof(half_t) + (T_ + A_ + A_ + 256) * sizeof(float);

  for (int step = 0; step < L_; ++step) {
    gemm(hh, sWh, sb, sProj, nullptr, B_, A_, S_);                        // sProj = h @ sW^T + sb
    attn_step<<<B_, 256, attn_lds, stream>>>(xPh, xh, sProj, wW, wb, emb, tgt, step, gin);
    gemm(gin, Wih_h, b_ih, gi, nullptr, B_, 3 * S_, A_ + DX_);            // gi = [yp,ctx] @ W_ih^T
    gemm(hh,  Whh_h, b_hh, gh, nullptr, B_, 3 * S_, S_);                  // gh = h @ W_hh^T
    gru_gate<<<(B_ * S_ + 255) / 256, 256, 0, stream>>>(gi, gh, h, hh);
    fc_out<<<(B_ * C_ + 255) / 256, 256, 0, stream>>>(h, fcW, fcb, out, step);
  }
}
